// BigramLanguageModel_51599737094908
// MI455X (gfx1250) — matrix operand order repaired and checked by reading, not compiled, not run on hardware
//
#include <hip/hip_runtime.h>
#include <hip/hip_bf16.h>
#include <math.h>

#define V_SZ 32000
#define T_SZ 1024
#define C_SZ 768
#define L_SZ 6
#define H_SZ 8
#define B_SZ 8
#define D_SZ (C_SZ / H_SZ)   /* 96 */
#define EPS_LN 1e-5f

typedef _Float16 v16h __attribute__((ext_vector_type(16)));
typedef _Float16 v8h  __attribute__((ext_vector_type(8)));
typedef _Float16 v4h  __attribute__((ext_vector_type(4)));
typedef float    v8f  __attribute__((ext_vector_type(8)));
typedef float    v4f  __attribute__((ext_vector_type(4)));

// ---------------------------------------------------------------------------
// Embedding: h[b,t,c] = tok_emb[x[b,t], c] + pos_emb[t, c]   (f32 residual)
// ---------------------------------------------------------------------------
__global__ void embed_kernel(const int* __restrict__ x,
                             const float* __restrict__ tok,
                             const float* __restrict__ pos,
                             float* __restrict__ h) {
  long idx = (long)blockIdx.x * blockDim.x + threadIdx.x;   // over B*T*C
  int  c   = (int)(idx % C_SZ);
  long bt  = idx / C_SZ;
  int  t   = (int)(bt % T_SZ);
  int  tokid = x[bt];
  h[idx] = tok[(long)tokid * C_SZ + c] + pos[(long)t * C_SZ + c];
}

// ---------------------------------------------------------------------------
// Transpose-cast weights: W[K][N] f32 -> Wt[N][K] f16 (K-contiguous rows so
// GEMM staging is pure 16B copies). 32x32 tile via LDS, 256 threads/block.
// ---------------------------------------------------------------------------
__global__ void transpose_cast_kernel(const float* __restrict__ W,
                                      _Float16* __restrict__ Wt,
                                      int K, int N) {
  __shared__ __align__(16) _Float16 t[32 * 40];
  const int tid = threadIdx.x;
  const int k0 = blockIdx.x * 32, n0 = blockIdx.y * 32;
  {
    int r = tid >> 5, c = tid & 31;           // coalesced reads along n
#pragma unroll
    for (int i = 0; i < 4; ++i) {
      int kk = r + i * 8;
      t[c * 40 + kk] = (_Float16)W[(long)(k0 + kk) * N + (n0 + c)];
    }
  }
  __syncthreads();
  {
    int c = tid >> 3, rq = (tid & 7) * 4;     // 8B vector writes along k
    v4h o = *(const v4h*)&t[c * 40 + rq];
    *(v4h*)&Wt[(long)(n0 + c) * K + (k0 + rq)] = o;
  }
}

// ---------------------------------------------------------------------------
// LayerNorm rows of C=768: f32 in (residual stream), f16 out (GEMM operand)
// ---------------------------------------------------------------------------
__global__ void layernorm_kernel(const float* __restrict__ in,
                                 const float* __restrict__ w,
                                 const float* __restrict__ b,
                                 _Float16* __restrict__ out) {
  __shared__ float red[256];
  const int tid = threadIdx.x;
  const float* row = in + (long)blockIdx.x * C_SZ;
  float v0 = row[tid], v1 = row[tid + 256], v2 = row[tid + 512];

  red[tid] = v0 + v1 + v2; __syncthreads();
  for (int s = 128; s > 0; s >>= 1) { if (tid < s) red[tid] += red[tid + s]; __syncthreads(); }
  float mean = red[0] * (1.0f / C_SZ);
  __syncthreads();

  float d0 = v0 - mean, d1 = v1 - mean, d2 = v2 - mean;
  red[tid] = d0 * d0 + d1 * d1 + d2 * d2; __syncthreads();
  for (int s = 128; s > 0; s >>= 1) { if (tid < s) red[tid] += red[tid + s]; __syncthreads(); }
  float rstd = rsqrtf(red[0] * (1.0f / C_SZ) + EPS_LN);

  _Float16* orow = out + (long)blockIdx.x * C_SZ;
  orow[tid]       = (_Float16)(d0 * rstd * w[tid]       + b[tid]);
  orow[tid + 256] = (_Float16)(d1 * rstd * w[tid + 256] + b[tid + 256]);
  orow[tid + 512] = (_Float16)(d2 * rstd * w[tid + 512] + b[tid + 512]);
}

// ---------------------------------------------------------------------------
// WMMA GEMM, f16 operands, f32 accumulate: D = A @ B [+bias][+res][ReLU]
//   8 waves (WAVES_M x WAVES_N); wave owns (MT*16) x 32 -> MT*2 wmma / K-step.
//   Double-buffered LDS tiles [rows][32] f16 (stride 40 halves, 16B-aligned
//   rows): fragments are ds_load_b128; staging uses CDNA5 async copies
//   (global_load_async_to_lds_b128, ASYNCcnt) so the K+1 tile copy overlaps
//   the K tile's WMMAs.  Pipeline per step:
//     s_wait_asynccnt 0 ; barrier ; async-prefetch k+32 into buf^1 ;
//     ds_load fragments from buf ; MT*2 x wmma.
//   B_TRANS: B element (k,n) at B[n*ldb + k] (transposed weights, K in Q@K^T)
//   CAUSAL : *scale then mask n>m -> -1e30; fully-masked tiles skip compute.
//   OUT_F16: store D as f16 (tensors consumed only by downstream GEMMs).
// ---------------------------------------------------------------------------
template<int WAVES_M, int WAVES_N, int MT, bool B_TRANS, bool CAUSAL,
         bool RELU, bool HAS_BIAS, bool HAS_RES, bool OUT_F16>
__global__ __launch_bounds__(256)
void gemm_wmma(const _Float16* __restrict__ A,  long lda, long a_so, long a_si,
               const _Float16* __restrict__ Bm, long ldb, long b_so, long b_si,
               void* __restrict__ Dmv,          long ldd, long d_so, long d_si,
               const float* __restrict__ bias,
               const float* __restrict__ Res,
               int K, int zi_count, float scale) {
  constexpr int BM = WAVES_M * MT * 16;
  constexpr int BN = WAVES_N * 32;
  constexpr int SA = 40;   // LDS row stride in halves (80B = multiple of 16B)
  constexpr int SB = 40;
  __shared__ __align__(16) _Float16 As[2 * BM * SA];   // double buffered
  __shared__ __align__(16) _Float16 Bs[2 * BN * SB];

  const int tid  = threadIdx.x;
  const int lane = tid & 31;
  const int wave = tid >> 5;
  const int wm   = wave % WAVES_M;
  const int wn   = wave / WAVES_M;

  const int z  = blockIdx.z;
  const int zo = z / zi_count;
  const int zi = z % zi_count;
  A  += (long)zo * a_so + (long)zi * a_si;
  Bm += (long)zo * b_so + (long)zi * b_si;
  const long doff = (long)zo * d_so + (long)zi * d_si;
  float*    Df = (float*)Dmv    + doff;
  _Float16* Dh = (_Float16*)Dmv + doff;

  const int rowBlock = blockIdx.y * BM;
  const int colBlock = blockIdx.x * BN;

  // ---- causal: tile entirely above diagonal -> constant fill, no compute ----
  if (CAUSAL && colBlock >= rowBlock + BM) {
#pragma unroll
    for (int mt = 0; mt < MT; ++mt)
#pragma unroll
      for (int nt = 0; nt < 2; ++nt) {
        const int n_local = colBlock + wn * 32 + nt * 16 + (lane & 15);
#pragma unroll
        for (int v = 0; v < 8; ++v) {
          const int m_local = rowBlock + (wm * MT + mt) * 16 + v + ((lane >> 4) << 3);
          Df[(long)m_local * ldd + n_local] = -1e30f;
        }
      }
    return;
  }

  // ---- async stage of one K-tile into LDS buffer `buf` ----
  auto stage_tiles = [&](int k0, int buf) {
    _Float16* Ab = &As[buf * BM * SA];
    _Float16* Bb = &Bs[buf * BN * SB];
    // A tile [BM][32]: per-lane 16B memory->LDS async copy
#pragma unroll
    for (int i = 0; i < (BM * 4) / 256; ++i) {
      int idx = tid + i * 256;                 // 8 halves per idx
      int m = idx >> 2, kq = (idx & 3) * 8;
      const _Float16* gp = &A[(long)(rowBlock + m) * lda + (k0 + kq)];
      unsigned lp = (unsigned)(size_t)&Ab[m * SA + kq];   // LDS = addr[31:0]
      asm volatile("global_load_async_to_lds_b128 %0, %1, off"
                   :: "v"(lp), "v"(gp) : "memory");
    }
    if (B_TRANS) {                             // K-contiguous source: async 16B
#pragma unroll
      for (int i = 0; i < (BN * 4) / 256; ++i) {
        int idx = tid + i * 256;
        int n = idx >> 2, kq = (idx & 3) * 8;
        const _Float16* gp = &Bm[(long)(colBlock + n) * ldb + (k0 + kq)];
        unsigned lp = (unsigned)(size_t)&Bb[n * SB + kq];
        asm volatile("global_load_async_to_lds_b128 %0, %1, off"
                     :: "v"(lp), "v"(gp) : "memory");
      }
    } else {                                   // N-contiguous source (attn@V only)
      constexpr int BLOADS = (BN * 32) / 8;
      if (BLOADS >= 256 || tid < BLOADS) {
        int kk = tid / (BN / 8), nq = (tid % (BN / 8)) * 8;
        v8h hv = *(const v8h*)&Bm[(long)(k0 + kk) * ldb + (colBlock + nq)];
#pragma unroll
        for (int j = 0; j < 8; ++j) Bb[(nq + j) * SB + kk] = hv[j];
      }
    }
  };

  v8f acc[MT][2];
#pragma unroll
  for (int mt = 0; mt < MT; ++mt) { acc[mt][0] = v8f{}; acc[mt][1] = v8f{}; }

  stage_tiles(0, 0);                           // prologue fill of buffer 0

  for (int k0 = 0, it = 0; k0 < K; k0 += 32, ++it) {
    const int cur = it & 1;
    asm volatile("s_wait_asynccnt 0" ::: "memory");  // this wave's copies done
    __syncthreads();                                 // all waves' copies visible;
                                                     // buf^1 fully consumed
    if (k0 + 32 < K) stage_tiles(k0 + 32, cur ^ 1);  // overlap copy with wmma

    const _Float16* Ab = &As[cur * BM * SA];
    const _Float16* Bb = &Bs[cur * BN * SB];

    // ---- B fragments (32x16): lane n = lane%16, K = (lane/16)*16 + h ----
    const int bkb = (lane >> 4) * 16;
    v16h bfrag[2];
#pragma unroll
    for (int nt = 0; nt < 2; ++nt) {
      const int bn = wn * 32 + nt * 16 + (lane & 15);
      v8h blo = *(const v8h*)&Bb[bn * SB + bkb];
      v8h bhi = *(const v8h*)&Bb[bn * SB + bkb + 8];
#pragma unroll
      for (int e = 0; e < 8; ++e) { bfrag[nt][e] = blo[e]; bfrag[nt][e + 8] = bhi[e]; }
    }

    // ---- A fragments (16x32) + MT*2 wmma ----
    const int akb = (lane >> 4) * 16;
#pragma unroll
    for (int mt = 0; mt < MT; ++mt) {
      const int am = (wm * MT + mt) * 16 + (lane & 15);
      v8h alo = *(const v8h*)&Ab[am * SA + akb];
      v8h ahi = *(const v8h*)&Ab[am * SA + akb + 8];
      v16h afrag;
#pragma unroll
      for (int e = 0; e < 8; ++e) { afrag[e] = alo[e]; afrag[e + 8] = ahi[e]; }
      acc[mt][0] = __builtin_amdgcn_wmma_f32_16x16x32_f16(
          false, afrag, false, bfrag[0], (short)0, acc[mt][0], false, false);
      acc[mt][1] = __builtin_amdgcn_wmma_f32_16x16x32_f16(
          false, afrag, false, bfrag[1], (short)0, acc[mt][1], false, false);
    }
  }

  // ---- epilogue ----
#pragma unroll
  for (int mt = 0; mt < MT; ++mt)
#pragma unroll
    for (int nt = 0; nt < 2; ++nt) {
      const int n_local = colBlock + wn * 32 + nt * 16 + (lane & 15);
      const float bv = HAS_BIAS ? bias[n_local] : 0.0f;
#pragma unroll
      for (int v = 0; v < 8; ++v) {
        const int m_local = rowBlock + (wm * MT + mt) * 16 + v + ((lane >> 4) << 3);
        float val = acc[mt][nt][v];
        if (CAUSAL) {
          val = (n_local <= m_local) ? val * scale : -1e30f;
        } else {
          val += bv;
          if (HAS_RES) val += Res[doff + (long)m_local * ldd + n_local];
          if (RELU)    val  = fmaxf(val, 0.0f);
        }
        if (OUT_F16) Dh[(long)m_local * ldd + n_local] = (_Float16)val;
        else         Df[(long)m_local * ldd + n_local] = val;
      }
    }
}

// ---------------------------------------------------------------------------
// Row softmax over T=1024: f32 scores in, f16 probabilities out
// ---------------------------------------------------------------------------
__global__ void softmax_kernel(const float* __restrict__ scores,
                               _Float16* __restrict__ probs) {
  __shared__ float red[256];
  const int tid = threadIdx.x;
  const float* row = scores + (long)blockIdx.x * T_SZ;
  _Float16*   orow = probs  + (long)blockIdx.x * T_SZ;
  float v[4];
  float mx = -3.4e38f;
#pragma unroll
  for (int i = 0; i < 4; ++i) { v[i] = row[tid + i * 256]; mx = fmaxf(mx, v[i]); }
  red[tid] = mx; __syncthreads();
  for (int s = 128; s > 0; s >>= 1) { if (tid < s) red[tid] = fmaxf(red[tid], red[tid + s]); __syncthreads(); }
  mx = red[0]; __syncthreads();
  float sum = 0.f;
#pragma unroll
  for (int i = 0; i < 4; ++i) { v[i] = expf(v[i] - mx); sum += v[i]; }
  red[tid] = sum; __syncthreads();
  for (int s = 128; s > 0; s >>= 1) { if (tid < s) red[tid] += red[tid + s]; __syncthreads(); }
  const float inv = 1.0f / red[0];
#pragma unroll
  for (int i = 0; i < 4; ++i) orow[tid + i * 256] = (_Float16)(v[i] * inv);
}

// ---------------------------------------------------------------------------
// Per-row log-softmax + NLL over V=32000 (one block per token)
// ---------------------------------------------------------------------------
__global__ void nll_kernel(const float* __restrict__ logits,
                           const int* __restrict__ y,
                           float* __restrict__ nll) {
  __shared__ float red[256];
  const int tid = threadIdx.x;
  const float* row = logits + (long)blockIdx.x * V_SZ;
  float mx = -3.4e38f;
  for (int i = tid; i < V_SZ; i += 256) mx = fmaxf(mx, row[i]);
  red[tid] = mx; __syncthreads();
  for (int s = 128; s > 0; s >>= 1) { if (tid < s) red[tid] = fmaxf(red[tid], red[tid + s]); __syncthreads(); }
  mx = red[0]; __syncthreads();
  float sum = 0.f;
  for (int i = tid; i < V_SZ; i += 256) sum += expf(row[i] - mx);
  red[tid] = sum; __syncthreads();
  for (int s = 128; s > 0; s >>= 1) { if (tid < s) red[tid] += red[tid + s]; __syncthreads(); }
  if (tid == 0) nll[blockIdx.x] = -(row[y[blockIdx.x]] - mx - logf(red[0]));
}

__global__ void loss_kernel(const float* __restrict__ nll, float* __restrict__ out) {
  __shared__ float red[256];
  const int tid = threadIdx.x;
  float s = 0.f;
  for (int i = tid; i < B_SZ * T_SZ; i += 256) s += nll[i];
  red[tid] = s; __syncthreads();
  for (int st = 128; st > 0; st >>= 1) { if (tid < st) red[tid] += red[tid + st]; __syncthreads(); }
  if (tid == 0) out[(long)B_SZ * T_SZ * V_SZ] = red[0] / (float)(B_SZ * T_SZ);
}

// ---------------------------------------------------------------------------
// Host: full forward pass
// ---------------------------------------------------------------------------
extern "C" void kernel_launch(void* const* d_in, const int* in_sizes, int n_in,
                              void* d_out, int out_size, void* d_ws, size_t ws_size,
                              hipStream_t stream) {
  (void)in_sizes; (void)n_in; (void)out_size; (void)ws_size;

  const int*   x       = (const int*)d_in[0];
  const int*   y       = (const int*)d_in[1];
  const float* tok_emb = (const float*)d_in[2];
  const float* pos_emb = (const float*)d_in[3];
  const float* ln1_w   = (const float*)d_in[4];
  const float* ln1_b   = (const float*)d_in[5];
  const float* attn_w  = (const float*)d_in[6];
  const float* attn_b  = (const float*)d_in[7];
  const float* proj_w  = (const float*)d_in[8];
  const float* proj_b  = (const float*)d_in[9];
  const float* ln2_w   = (const float*)d_in[10];
  const float* ln2_b   = (const float*)d_in[11];
  const float* ffn_w1  = (const float*)d_in[12];
  const float* ffn_b1  = (const float*)d_in[13];
  const float* ffn_w2  = (const float*)d_in[14];
  const float* ffn_b2  = (const float*)d_in[15];
  const float* lnf_w   = (const float*)d_in[16];
  const float* lnf_b   = (const float*)d_in[17];
  const float* head_w  = (const float*)d_in[18];
  const float* head_b  = (const float*)d_in[19];

  const long BT  = (long)B_SZ * T_SZ;                  // 8192 rows
  const long BHT = (long)B_SZ * H_SZ * T_SZ * T_SZ;    // scores elements

  char* wsb = (char*)d_ws;
  float*     h      = (float*)wsb;      wsb += BT * C_SZ * 4;        // residual
  _Float16*  hn     = (_Float16*)wsb;   wsb += BT * C_SZ * 2;        // LN out
  _Float16*  qkv    = (_Float16*)wsb;   wsb += BT * 3 * C_SZ * 2;
  float*     scores = (float*)wsb;      wsb += BHT * 4;
  _Float16*  probs  = (_Float16*)wsb;   wsb += BHT * 2;
  _Float16*  aout   = (_Float16*)wsb;   wsb += BT * C_SZ * 2;
  _Float16*  ffn1   = (_Float16*)wsb;   wsb += BT * 4 * C_SZ * 2;
  float*     nll    = (float*)wsb;      wsb += BT * 4;
  _Float16*  wts    = (_Float16*)wsb;   // transposed f16 weight arena
  const long WPL = (long)C_SZ * 3 * C_SZ + (long)C_SZ * C_SZ
                 + (long)C_SZ * 4 * C_SZ + (long)4 * C_SZ * C_SZ;    // per layer
  _Float16* headT = wts + (long)L_SZ * WPL;
  float* logits = (float*)d_out;                       // [B,T,V] then loss

  const float scale = 0.10206207261596575f;            // D^-0.5, D=96

  // ---- prologue: transpose-cast all weights to f16 [N][K] ----
  for (int l = 0; l < L_SZ; ++l) {
    _Float16* awT = wts + (long)l * WPL;
    _Float16* pwT = awT + (long)C_SZ * 3 * C_SZ;
    _Float16* w1T = pwT + (long)C_SZ * C_SZ;
    _Float16* w2T = w1T + (long)C_SZ * 4 * C_SZ;
    transpose_cast_kernel<<<dim3(C_SZ / 32, 3 * C_SZ / 32), 256, 0, stream>>>(
        attn_w + (long)l * C_SZ * 3 * C_SZ, awT, C_SZ, 3 * C_SZ);
    transpose_cast_kernel<<<dim3(C_SZ / 32, C_SZ / 32), 256, 0, stream>>>(
        proj_w + (long)l * C_SZ * C_SZ, pwT, C_SZ, C_SZ);
    transpose_cast_kernel<<<dim3(C_SZ / 32, 4 * C_SZ / 32), 256, 0, stream>>>(
        ffn_w1 + (long)l * C_SZ * 4 * C_SZ, w1T, C_SZ, 4 * C_SZ);
    transpose_cast_kernel<<<dim3(4 * C_SZ / 32, C_SZ / 32), 256, 0, stream>>>(
        ffn_w2 + (long)l * 4 * C_SZ * C_SZ, w2T, 4 * C_SZ, C_SZ);
  }
  transpose_cast_kernel<<<dim3(C_SZ / 32, V_SZ / 32), 256, 0, stream>>>(
      head_w, headT, C_SZ, V_SZ);

  // h = tok_emb[x] + pos_emb
  embed_kernel<<<(BT * C_SZ) / 256, 256, 0, stream>>>(x, tok_emb, pos_emb, h);

  for (int l = 0; l < L_SZ; ++l) {
    _Float16* awT = wts + (long)l * WPL;
    _Float16* pwT = awT + (long)C_SZ * 3 * C_SZ;
    _Float16* w1T = pwT + (long)C_SZ * C_SZ;
    _Float16* w2T = w1T + (long)C_SZ * 4 * C_SZ;
    const float* lw1 = ln1_w  + (long)l * C_SZ;
    const float* lb1 = ln1_b  + (long)l * C_SZ;
    const float* ab  = attn_b + (long)l * 3 * C_SZ;
    const float* pb  = proj_b + (long)l * C_SZ;
    const float* lw2 = ln2_w  + (long)l * C_SZ;
    const float* lb2 = ln2_b  + (long)l * C_SZ;
    const float* b1  = ffn_b1 + (long)l * 4 * C_SZ;
    const float* b2  = ffn_b2 + (long)l * C_SZ;

    // hn = LN1(h)                                (f16 out)
    layernorm_kernel<<<BT, 256, 0, stream>>>(h, lw1, lb1, hn);

    // qkv = hn @ aw + ab                         [8192,768]@[768,2304], f16 out
    gemm_wmma<2, 4, 2, true, false, false, true, false, true>
        <<<dim3(3 * C_SZ / 128, BT / 64, 1), 256, 0, stream>>>(
        hn, C_SZ, 0, 0,  awT, C_SZ, 0, 0,  qkv, 3 * C_SZ, 0, 0,
        ab, nullptr, C_SZ, 1, 0.f);

    // scores[b,h] = scale * q @ k^T, causal      (z = b*H + h), f32 out
    gemm_wmma<2, 4, 2, true, true, false, false, false, false>
        <<<dim3(T_SZ / 128, T_SZ / 64, B_SZ * H_SZ), 256, 0, stream>>>(
        qkv,         3 * C_SZ, (long)T_SZ * 3 * C_SZ, D_SZ,       // q
        qkv + C_SZ,  3 * C_SZ, (long)T_SZ * 3 * C_SZ, D_SZ,       // k
        scores, T_SZ, (long)H_SZ * T_SZ * T_SZ, (long)T_SZ * T_SZ,
        nullptr, nullptr, D_SZ, H_SZ, scale);

    // probs = softmax(scores)                    (f16 out)
    softmax_kernel<<<B_SZ * H_SZ * T_SZ, 256, 0, stream>>>(scores, probs);

    // aout[b,t,h*D+d] = probs[b,h] @ v[b,h]      [1024,1024]@[1024,96], f16 out
    gemm_wmma<8, 1, 1, false, false, false, false, false, true>
        <<<dim3(D_SZ / 32, T_SZ / 128, B_SZ * H_SZ), 256, 0, stream>>>(
        probs, T_SZ, (long)H_SZ * T_SZ * T_SZ, (long)T_SZ * T_SZ,
        qkv + 2 * C_SZ, 3 * C_SZ, (long)T_SZ * 3 * C_SZ, D_SZ,    // v
        aout, C_SZ, (long)T_SZ * C_SZ, D_SZ,
        nullptr, nullptr, T_SZ, H_SZ, 0.f);

    // h = h + aout @ pw + pb                     [8192,768]@[768,768], f32 out
    gemm_wmma<2, 4, 2, true, false, false, true, true, false>
        <<<dim3(C_SZ / 128, BT / 64, 1), 256, 0, stream>>>(
        aout, C_SZ, 0, 0,  pwT, C_SZ, 0, 0,  h, C_SZ, 0, 0,
        pb, h, C_SZ, 1, 0.f);

    // hn = LN2(h)
    layernorm_kernel<<<BT, 256, 0, stream>>>(h, lw2, lb2, hn);

    // ffn1 = relu(hn @ w1 + b1)                  [8192,768]@[768,3072], f16 out
    gemm_wmma<2, 4, 2, true, false, true, true, false, true>
        <<<dim3(4 * C_SZ / 128, BT / 64, 1), 256, 0, stream>>>(
        hn, C_SZ, 0, 0,  w1T, C_SZ, 0, 0,  ffn1, 4 * C_SZ, 0, 0,
        b1, nullptr, C_SZ, 1, 0.f);

    // h = h + ffn1 @ w2 + b2                     [8192,3072]@[3072,768], f32 out
    gemm_wmma<2, 4, 2, true, false, false, true, true, false>
        <<<dim3(C_SZ / 128, BT / 64, 1), 256, 0, stream>>>(
        ffn1, 4 * C_SZ, 0, 0,  w2T, 4 * C_SZ, 0, 0,  h, C_SZ, 0, 0,
        b2, h, 4 * C_SZ, 1, 0.f);
  }

  // hn = LN_f(h)
  layernorm_kernel<<<BT, 256, 0, stream>>>(h, lnf_w, lnf_b, hn);

  // logits = hn @ head_w + head_b                [8192,768]@[768,32000]
  gemm_wmma<2, 4, 2, true, false, false, true, false, false>
      <<<dim3(V_SZ / 128, BT / 64, 1), 256, 0, stream>>>(
      hn, C_SZ, 0, 0,  headT, C_SZ, 0, 0,  logits, V_SZ, 0, 0,
      head_b, nullptr, C_SZ, 1, 0.f);

  // loss = mean(-log_softmax(logits)[y])
  nll_kernel<<<BT, 256, 0, stream>>>(logits, y, nll);
  loss_kernel<<<1, 256, 0, stream>>>(nll, (float*)d_out);
}
